// model002_37666863186443
// MI455X (gfx1250) — compile-verified, matrix operand order checked by reading, not run on hardware
//
#include <hip/hip_runtime.h>

typedef __attribute__((ext_vector_type(16))) __bf16 v16bf;
typedef __attribute__((ext_vector_type(8)))  __bf16 v8bf;
typedef __attribute__((ext_vector_type(8)))  float  v8f;

#define NY   64
#define NU   64
#define NHID 512
#define NHRZ 1024
#define NB   64
#define CCH  64      // chunks
#define LCH  16      // timesteps per chunk (NHRZ = CCH*LCH)

// ---------------------------------------------------------------------------
// helpers: build 16-bit WMMA operands.  Lane L -> row L%16; its K values in a
// K=32 window are {kb..kb+7} u {kb+16..kb+23}, kb = 8*(L/16) (+k0).
// ---------------------------------------------------------------------------
__device__ __forceinline__ v16bf mk_a_f32(const float* __restrict__ row, int kb)
{
    v16bf a;
#pragma unroll
    for (int e = 0; e < 8; ++e) {
        a[e]     = (__bf16)row[kb + e];
        a[e + 8] = (__bf16)row[kb + 16 + e];
    }
    return a;
}

__device__ __forceinline__ v16bf mk_b_bf16(const __bf16* row, int kb)
{
    const v8bf lo = *(const v8bf*)(row + kb);
    const v8bf hi = *(const v8bf*)(row + kb + 16);
    v16bf b;
#pragma unroll
    for (int e = 0; e < 8; ++e) { b[e] = lo[e]; b[e + 8] = hi[e]; }
    return b;
}

__device__ __forceinline__ v8f wmma_bf16(v16bf a, v16bf b, v8f c)
{
    return __builtin_amdgcn_wmma_f32_16x16x32_bf16(
        false, a, false, b, (short)0, c, false, false);
}

// ---------------------------------------------------------------------------
// Weight f32 -> bf16 conversion.
// ---------------------------------------------------------------------------
__global__ __launch_bounds__(256)
void k_cvt(const float* __restrict__ src, __bf16* __restrict__ dst, int n)
{
    const int i = blockIdx.x * 256 + threadIdx.x;
    if (i < n) dst[i] = (__bf16)src[i];
}

// ---------------------------------------------------------------------------
// Pass 1: per-chunk carry = sum_{s<L} R^{L-1-s} bu[cL+s]  (zero-seeded scan).
// Block = (chunk c, batch tile bt).  Wave w owns h-tiles {2w,2w+1,2w+16,2w+17}
// so each rotation pair (h, h+256) is lane/VGPR-aligned across two acc tiles.
// ---------------------------------------------------------------------------
__global__ __launch_bounds__(256)
void k_carry(const float* __restrict__ U, const __bf16* __restrict__ Bbf,
             const float* __restrict__ lr_, const float* __restrict__ li_,
             float* __restrict__ carry)
{
    const int c    = blockIdx.x;
    const int bt   = blockIdx.y;
    const int wave = threadIdx.x >> 5;
    const int lane = threadIdx.x & 31;
    const int lm   = lane & 15;
    const int kh   = lane >> 4;
    const int ht0  = 2 * wave, ht1 = 2 * wave + 1;

    // B operands: invariant over t, keep in registers.
    const v16bf B00 = mk_b_bf16(Bbf + (ht0 * 16 + lm) * NU, 8 * kh);
    const v16bf B01 = mk_b_bf16(Bbf + (ht0 * 16 + lm) * NU, 32 + 8 * kh);
    const v16bf B10 = mk_b_bf16(Bbf + (ht1 * 16 + lm) * NU, 8 * kh);
    const v16bf B11 = mk_b_bf16(Bbf + (ht1 * 16 + lm) * NU, 32 + 8 * kh);
    const v16bf B20 = mk_b_bf16(Bbf + ((ht0 + 16) * 16 + lm) * NU, 8 * kh);
    const v16bf B21 = mk_b_bf16(Bbf + ((ht0 + 16) * 16 + lm) * NU, 32 + 8 * kh);
    const v16bf B30 = mk_b_bf16(Bbf + ((ht1 + 16) * 16 + lm) * NU, 8 * kh);
    const v16bf B31 = mk_b_bf16(Bbf + ((ht1 + 16) * 16 + lm) * NU, 32 + 8 * kh);

    const float lr0 = lr_[32 * wave + lm],      li0 = li_[32 * wave + lm];
    const float lr1 = lr_[32 * wave + 16 + lm], li1 = li_[32 * wave + 16 + lm];

    v8f s0 = {}, s1 = {}, s2 = {}, s3 = {};

#pragma unroll 1
    for (int tt = 0; tt < LCH; ++tt) {
        const int t = c * LCH + tt;
        const float* arow = U + (size_t)t * (NB * NU) + (bt * 16 + lm) * NU;
        const v16bf a0 = mk_a_f32(arow, 8 * kh);
        const v16bf a1 = mk_a_f32(arow, 32 + 8 * kh);

        v8f d0 = {}, d1 = {}, d2 = {}, d3 = {};
        d0 = wmma_bf16(a0, B00, d0); d0 = wmma_bf16(a1, B01, d0);
        d1 = wmma_bf16(a0, B10, d1); d1 = wmma_bf16(a1, B11, d1);
        d2 = wmma_bf16(a0, B20, d2); d2 = wmma_bf16(a1, B21, d2);
        d3 = wmma_bf16(a0, B30, d3); d3 = wmma_bf16(a1, B31, d3);

#pragma unroll
        for (int r = 0; r < 8; ++r) {
            const float n1 = fmaf(lr0, s0[r], fmaf(-li0, s2[r], d0[r]));
            const float n2 = fmaf(li0, s0[r], fmaf( lr0, s2[r], d2[r]));
            const float m1 = fmaf(lr1, s1[r], fmaf(-li1, s3[r], d1[r]));
            const float m2 = fmaf(li1, s1[r], fmaf( lr1, s3[r], d3[r]));
            s0[r] = n1; s2[r] = n2; s1[r] = m1; s3[r] = m2;
        }
    }

    float* cr = carry + ((size_t)c * NB + bt * 16) * NHID;
#pragma unroll
    for (int r = 0; r < 8; ++r) {
        const int row = r + 8 * kh;
        cr[row * NHID + ht0 * 16 + lm]        = s0[r];
        cr[row * NHID + ht1 * 16 + lm]        = s1[r];
        cr[row * NHID + (ht0 + 16) * 16 + lm] = s2[r];
        cr[row * NHID + (ht1 + 16) * 16 + lm] = s3[r];
    }
}

// ---------------------------------------------------------------------------
// Combine: xs[0] = x0;  xs[c+1] = R^L xs[c] + carry[c].  Thread per (b, pair).
// ---------------------------------------------------------------------------
__global__ __launch_bounds__(256)
void k_combine(const float* __restrict__ y0, const float* __restrict__ Wy2x,
               const float* __restrict__ by2x, const float* __restrict__ lr_,
               const float* __restrict__ li_, const float* __restrict__ carry,
               float* __restrict__ xs)
{
    const int tid = blockIdx.x * 256 + threadIdx.x; // 0..16383
    const int b   = tid >> 8;
    const int j   = tid & 255;

    // x0 = y0 @ Wy2x^T + by2x
    float x1 = by2x[j], x2 = by2x[j + 256];
    const float* yrow = y0 + b * NY;
    const float* w1   = Wy2x + (size_t)j * NY;
    const float* w2   = Wy2x + (size_t)(j + 256) * NY;
#pragma unroll 8
    for (int u = 0; u < NY; ++u) {
        const float yv = yrow[u];
        x1 = fmaf(yv, w1[u], x1);
        x2 = fmaf(yv, w2[u], x2);
    }

    // lambda^LCH (LCH = 16 -> 4 complex squarings)
    float ar = lr_[j], ai = li_[j];
#pragma unroll
    for (int q = 0; q < 4; ++q) {
        const float nr = ar * ar - ai * ai;
        const float ni = 2.0f * ar * ai;
        ar = nr; ai = ni;
    }

#pragma unroll 1
    for (int c = 0; c < CCH; ++c) {
        float* xsc = xs + ((size_t)c * NB + b) * NHID + j;
        xsc[0]   = x1;
        xsc[256] = x2;
        const float* crc = carry + ((size_t)c * NB + b) * NHID + j;
        const float c1 = crc[0], c2 = crc[256];
        const float n1 = fmaf(ar, x1, fmaf(-ai, x2, c1));
        const float n2 = fmaf(ai, x1, fmaf( ar, x2, c2));
        x1 = n1; x2 = n2;
    }
}

// ---------------------------------------------------------------------------
// Pass 2: recompute bu, run register scan seeded from xs[c], and per timestep
// do the output GEMM through LDS:  Y[t] = X[t] @ Wx2y^T + bx2y.
// Output GEMM: wave w -> y-tile (w&3), K half (w>>2); partial reduced in LDS.
// ---------------------------------------------------------------------------
__global__ __launch_bounds__(256)
void k_fused(const float* __restrict__ U, const __bf16* __restrict__ Bbf,
             const __bf16* __restrict__ Wybf, const float* __restrict__ lr_,
             const float* __restrict__ li_, const float* __restrict__ xs,
             const float* __restrict__ bx2y, float* __restrict__ Y)
{
    __shared__ __bf16 Xl[16 * NHID];   // 16 KB: X[t] tile, bf16
    __shared__ float  Pl[4 * 256];     // 4 KB: K-split partial sums

    const int c    = blockIdx.x;
    const int bt   = blockIdx.y;
    const int wave = threadIdx.x >> 5;
    const int lane = threadIdx.x & 31;
    const int lm   = lane & 15;
    const int kh   = lane >> 4;
    const int ht0  = 2 * wave, ht1 = 2 * wave + 1;

    const v16bf B00 = mk_b_bf16(Bbf + (ht0 * 16 + lm) * NU, 8 * kh);
    const v16bf B01 = mk_b_bf16(Bbf + (ht0 * 16 + lm) * NU, 32 + 8 * kh);
    const v16bf B10 = mk_b_bf16(Bbf + (ht1 * 16 + lm) * NU, 8 * kh);
    const v16bf B11 = mk_b_bf16(Bbf + (ht1 * 16 + lm) * NU, 32 + 8 * kh);
    const v16bf B20 = mk_b_bf16(Bbf + ((ht0 + 16) * 16 + lm) * NU, 8 * kh);
    const v16bf B21 = mk_b_bf16(Bbf + ((ht0 + 16) * 16 + lm) * NU, 32 + 8 * kh);
    const v16bf B30 = mk_b_bf16(Bbf + ((ht1 + 16) * 16 + lm) * NU, 8 * kh);
    const v16bf B31 = mk_b_bf16(Bbf + ((ht1 + 16) * 16 + lm) * NU, 32 + 8 * kh);

    const float lr0 = lr_[32 * wave + lm],      li0 = li_[32 * wave + lm];
    const float lr1 = lr_[32 * wave + 16 + lm], li1 = li_[32 * wave + 16 + lm];

    // seed state from chunk start-state xs[c]
    v8f s0, s1, s2, s3;
    {
        const float* xsc = xs + ((size_t)c * NB + bt * 16) * NHID;
#pragma unroll
        for (int r = 0; r < 8; ++r) {
            const int row = r + 8 * kh;
            s0[r] = xsc[row * NHID + ht0 * 16 + lm];
            s1[r] = xsc[row * NHID + ht1 * 16 + lm];
            s2[r] = xsc[row * NHID + (ht0 + 16) * 16 + lm];
            s3[r] = xsc[row * NHID + (ht1 + 16) * 16 + lm];
        }
    }

    const int yt    = wave & 3;       // output tile for GEMM phase
    const int khalf = wave >> 2;      // K half: 0 -> [0,256), 1 -> [256,512)
    const __bf16* brow = Wybf + (yt * 16 + lm) * NHID + khalf * 256;
    const float   bv   = bx2y[yt * 16 + lm];

#pragma unroll 1
    for (int tt = 0; tt < LCH; ++tt) {
        const int t = c * LCH + tt;
        const float* arow = U + (size_t)t * (NB * NU) + (bt * 16 + lm) * NU;
        const v16bf a0 = mk_a_f32(arow, 8 * kh);
        const v16bf a1 = mk_a_f32(arow, 32 + 8 * kh);

        v8f d0 = {}, d1 = {}, d2 = {}, d3 = {};
        d0 = wmma_bf16(a0, B00, d0); d0 = wmma_bf16(a1, B01, d0);
        d1 = wmma_bf16(a0, B10, d1); d1 = wmma_bf16(a1, B11, d1);
        d2 = wmma_bf16(a0, B20, d2); d2 = wmma_bf16(a1, B21, d2);
        d3 = wmma_bf16(a0, B30, d3); d3 = wmma_bf16(a1, B31, d3);

#pragma unroll
        for (int r = 0; r < 8; ++r) {
            const float n1 = fmaf(lr0, s0[r], fmaf(-li0, s2[r], d0[r]));
            const float n2 = fmaf(li0, s0[r], fmaf( lr0, s2[r], d2[r]));
            const float m1 = fmaf(lr1, s1[r], fmaf(-li1, s3[r], d1[r]));
            const float m2 = fmaf(li1, s1[r], fmaf( lr1, s3[r], d3[r]));
            s0[r] = n1; s2[r] = n2; s1[r] = m1; s3[r] = m2;
        }

        // X[t] tile -> LDS (bf16)
#pragma unroll
        for (int r = 0; r < 8; ++r) {
            const int row = r + 8 * kh;
            Xl[row * NHID + ht0 * 16 + lm]        = (__bf16)s0[r];
            Xl[row * NHID + ht1 * 16 + lm]        = (__bf16)s1[r];
            Xl[row * NHID + (ht0 + 16) * 16 + lm] = (__bf16)s2[r];
            Xl[row * NHID + (ht1 + 16) * 16 + lm] = (__bf16)s3[r];
        }
        __syncthreads();

        // output GEMM: A from LDS (row = lm), B = Wx2y bf16 (cached)
        const __bf16* al = Xl + lm * NHID + khalf * 256;
        v8f p = {};
#pragma unroll
        for (int k0 = 0; k0 < 256; k0 += 32) {
            const int kb = k0 + 8 * kh;
            const v16bf a = mk_b_bf16(al, kb);
            const v16bf b = mk_b_bf16(brow, kb);
            p = wmma_bf16(a, b, p);
        }

        if (wave >= 4) {
#pragma unroll
            for (int r = 0; r < 8; ++r) Pl[yt * 256 + r * 32 + lane] = p[r];
        }
        __syncthreads();

        if (wave < 4) {
            float* out = Y + (size_t)t * (NB * NY);
#pragma unroll
            for (int r = 0; r < 8; ++r) {
                const float v = p[r] + Pl[yt * 256 + r * 32 + lane] + bv;
                out[(bt * 16 + r + 8 * kh) * NY + yt * 16 + lm] = v;
            }
        }
        __syncthreads();   // protect Xl/Pl before next timestep
    }
}

// ---------------------------------------------------------------------------
extern "C" void kernel_launch(void* const* d_in, const int* in_sizes, int n_in,
                              void* d_out, int out_size, void* d_ws, size_t ws_size,
                              hipStream_t stream)
{
    const float* y0   = (const float*)d_in[0];
    const float* U    = (const float*)d_in[1];
    const float* lr   = (const float*)d_in[2];
    const float* li   = (const float*)d_in[3];
    const float* Bm   = (const float*)d_in[4];
    const float* Wy2x = (const float*)d_in[5];
    const float* by2x = (const float*)d_in[6];
    const float* Wx2y = (const float*)d_in[7];
    const float* bx2y = (const float*)d_in[8];

    float* Y = (float*)d_out;

    // ws: [Bbf 64KB][Wybf 64KB][carry 8MB][xs 8MB]  (~16.9 MB total)
    __bf16* Bbf   = (__bf16*)d_ws;
    __bf16* Wybf  = Bbf + NHID * NU;
    float*  carry = (float*)((char*)d_ws + 2 * NHID * NU * 2);
    float*  xs    = carry + (size_t)CCH * NB * NHID;

    k_cvt    <<<dim3(128),     256, 0, stream>>>(Bm,   Bbf,  NHID * NU);
    k_cvt    <<<dim3(128),     256, 0, stream>>>(Wx2y, Wybf, NY * NHID);
    k_carry  <<<dim3(CCH, 4),  256, 0, stream>>>(U, Bbf, lr, li, carry);
    k_combine<<<dim3(64),      256, 0, stream>>>(y0, Wy2x, by2x, lr, li, carry, xs);
    k_fused  <<<dim3(CCH, 4),  256, 0, stream>>>(U, Bbf, Wybf, lr, li, xs, bx2y, Y);
}
